// GaussianNeuralNetworkConv_26852135535196
// MI455X (gfx1250) — matrix operand order repaired and checked by reading, not compiled, not run on hardware
//
#include <hip/hip_runtime.h>

#define B_    4
#define N_    4096
#define CIN_  128
#define COUT_ 128
#define KT_STRIDE 36   // padded dword stride of a 64-col bf16 kernel-tile row

typedef __bf16 v16bf __attribute__((ext_vector_type(16)));
typedef float  v8f   __attribute__((ext_vector_type(8)));
typedef unsigned int v8u __attribute__((ext_vector_type(8)));

// RNE f32->bf16 pair, merged with one v_perm_b32: result {bf16(hi), bf16(lo)}
static __device__ __forceinline__ unsigned pack_bf16x2(float lo, float hi) {
    unsigned ul = __builtin_bit_cast(unsigned, lo);
    unsigned uh = __builtin_bit_cast(unsigned, hi);
    ul += 0x7fffu + ((ul >> 16) & 1u);
    uh += 0x7fffu + ((uh >> 16) & 1u);
    return __builtin_amdgcn_perm(uh, ul, 0x07060302u);  // {uh[31:16], ul[31:16]}
}

// ---------------------------------------------------------------------------
// Kernel 0: f32 -> bf16 (pairwise, dword stores). Used for x and W.
// ---------------------------------------------------------------------------
__global__ void cvt_bf16_kernel(const float* __restrict__ src,
                                unsigned* __restrict__ dst, int npairs) {
    int i = blockIdx.x * blockDim.x + threadIdx.x;
    if (i < npairs) {
        float2 f = ((const float2*)src)[i];
        dst[i] = pack_bf16x2(f.x, f.y);
    }
}

// ---------------------------------------------------------------------------
// Kernel 1: x_proj[b,n,o] = sum_i x[b,n,i]*W[o,i] + bias[o]
// Block = 16-row tile; 8 waves, wave w owns o-tile w. All A/B loads are b128
// from L2-resident bf16 copies. Output stored TRANSPOSED: xpT[b][o][n] (bf16).
// ---------------------------------------------------------------------------
__global__ __launch_bounds__(256) void proj_kernel(
        const unsigned short* __restrict__ xb,   // bf16 x, row-major [b][n][k]
        const unsigned short* __restrict__ Wb,   // bf16 W, row-major [o][k]
        const float* __restrict__ bias,
        unsigned short* __restrict__ xpT) {
    const int bidx  = blockIdx.x >> 8;      // N_/16 = 256 tiles per batch
    const int ntile = blockIdx.x & 255;
    const int n0    = ntile * 16;
    const int wave  = threadIdx.x >> 5;
    const int lane  = threadIdx.x & 31;
    const int M     = lane & 15;            // A row / D column id within tile
    const int hi    = lane >> 4;
    const int o     = wave * 16 + M;        // output channel (B/D column)

    const uint4* xa  = (const uint4*)(xb + ((size_t)bidx * N_ + (size_t)(n0 + M)) * CIN_);
    const uint4* wb4 = (const uint4*)Wb;

    v8f acc = {};
#pragma unroll
    for (int c = 0; c < 4; ++c) {
        uint4 a0 = xa[c * 4 + hi];
        uint4 a1 = xa[c * 4 + 2 + hi];
        uint4 b0 = wb4[(size_t)o * 16 + c * 4 + hi];
        uint4 b1 = wb4[(size_t)o * 16 + c * 4 + 2 + hi];
        v8u abits, bbits;
        abits[0] = a0.x; abits[1] = a0.y; abits[2] = a0.z; abits[3] = a0.w;
        abits[4] = a1.x; abits[5] = a1.y; abits[6] = a1.z; abits[7] = a1.w;
        bbits[0] = b0.x; bbits[1] = b0.y; bbits[2] = b0.z; bbits[3] = b0.w;
        bbits[4] = b1.x; bbits[5] = b1.y; bbits[6] = b1.z; bbits[7] = b1.w;
        v16bf a  = __builtin_bit_cast(v16bf, abits);
        v16bf bm = __builtin_bit_cast(v16bf, bbits);
        acc = __builtin_amdgcn_wmma_f32_16x16x32_bf16(
                  false, a, false, bm, (short)0, acc, false, false);
    }

    const float bv = bias[o];
    unsigned p0 = pack_bf16x2(acc[0] + bv, acc[1] + bv);
    unsigned p1 = pack_bf16x2(acc[2] + bv, acc[3] + bv);
    unsigned p2 = pack_bf16x2(acc[4] + bv, acc[5] + bv);
    unsigned p3 = pack_bf16x2(acc[6] + bv, acc[7] + bv);
    size_t base = ((size_t)bidx * COUT_ + o) * N_ + n0 + hi * 8;
    *(uint4*)(xpT + base) = make_uint4(p0, p1, p2, p3);
}

// ---------------------------------------------------------------------------
// Kernel 2: fused Gaussian conv. K-chunk = 64 (2 WMMAs) per iteration with
// LDS double buffering -> 1 barrier per 2 WMMAs. 9 waves: waves 0-7 own
// 16-wide o-tiles; wave 8 uses an all-ones B so its accumulator is the
// row sum (the L1 normalizer).
// ---------------------------------------------------------------------------
__global__ __launch_bounds__(288) void gconv_kernel(
        const float* __restrict__ coords, const unsigned short* __restrict__ xpT,
        const float* __restrict__ sigma, float* __restrict__ out) {
    __shared__ float    s_cx[16], s_cy[16];
    __shared__ float    s_rowsum[16];
    __shared__ unsigned s_ktile[2][16 * KT_STRIDE];  // 16 rows x 64 bf16 cols

    const int bidx  = blockIdx.x >> 8;
    const int ntile = blockIdx.x & 255;
    const int n0    = ntile * 16;
    const int t     = threadIdx.x;
    const int wave  = t >> 5;
    const int lane  = t & 31;
    const int M     = lane & 15;
    const int hi    = lane >> 4;

    const float s       = sigma[0];
    const float neg_inv = -1.0f / (2.0f * s * s);
    const float2* c2    = (const float2*)coords + (size_t)bidx * N_;

    if (t < 16) {
        float2 c = c2[n0 + t];
        s_cx[t] = c.x;
        s_cy[t] = c.y;
    }
    __syncthreads();

    float my_cx = 0.f, my_cy = 0.f;
    int   my_n = 0, my_d = 0;
    if (t < 256) {                 // thread computes 4 kernel entries (2 dwords)
        my_n = t >> 4;             // tile row 0..15
        my_d = t & 15;             // dword (m-pair) 0..15 within each 32-chunk
        my_cx = s_cx[my_n];
        my_cy = s_cy[my_n];
    }

    v8f acc = {};
    v8u ones;
#pragma unroll
    for (int v = 0; v < 8; ++v) ones[v] = 0x3F803F80u;   // bf16 1.0 x2

    const unsigned* brow = nullptr;
    if (wave < 8) {
        int o = wave * 16 + M;
        brow = (const unsigned*)(xpT + ((size_t)bidx * COUT_ + o) * N_);
    }

    // Tile generator: 64 columns starting at m0 into buffer `buf`.
    auto gen_tile = [&](int m0, int buf) {
        unsigned* row = &s_ktile[buf][my_n * KT_STRIDE];
#pragma unroll
        for (int h = 0; h < 2; ++h) {
            int m = m0 + h * 32 + my_d * 2;
            float2 ca = c2[m];
            float2 cb = c2[m + 1];
            float dx0 = my_cx - ca.x, dy0 = my_cy - ca.y;
            float dx1 = my_cx - cb.x, dy1 = my_cy - cb.y;
            float k0 = __expf((dx0 * dx0 + dy0 * dy0) * neg_inv);
            float k1 = __expf((dx1 * dx1 + dy1 * dy1) * neg_inv);
            row[h * 16 + my_d] = pack_bf16x2(k0, k1);
        }
    };

    if (t < 256) gen_tile(0, 0);     // prologue
    __syncthreads();

    const int NIT = N_ / 64;
    for (int it = 0; it < NIT; ++it) {
        const int m0  = it * 64;
        const int cur = it & 1;

        if (it + 1 < NIT && t < 256) gen_tile(m0 + 64, cur ^ 1);

#pragma unroll
        for (int sub = 0; sub < 2; ++sub) {   // two K=32 WMMAs
            const unsigned* krow = &s_ktile[cur][M * KT_STRIDE + sub * 16];
            uint4 qa0 = *(const uint4*)(krow + hi * 8);
            uint4 qa1 = *(const uint4*)(krow + hi * 8 + 4);
            v8u abits;
            abits[0] = qa0.x; abits[1] = qa0.y; abits[2] = qa0.z; abits[3] = qa0.w;
            abits[4] = qa1.x; abits[5] = qa1.y; abits[6] = qa1.z; abits[7] = qa1.w;
            v16bf a = __builtin_bit_cast(v16bf, abits);

            v8u bbits;
            if (wave < 8) {
                const uint4* bp =
                    (const uint4*)(brow + ((m0 + sub * 32) >> 1) + hi * 8);
                uint4 q0 = bp[0];
                uint4 q1 = bp[1];
                bbits[0] = q0.x; bbits[1] = q0.y; bbits[2] = q0.z; bbits[3] = q0.w;
                bbits[4] = q1.x; bbits[5] = q1.y; bbits[6] = q1.z; bbits[7] = q1.w;
                if (sub == 0)
                    __builtin_prefetch(brow + ((m0 + 64) >> 1) + hi * 8, 0, 0);
            } else {
                bbits = ones;               // rowsum wave
            }
            v16bf bm = __builtin_bit_cast(v16bf, bbits);

            acc = __builtin_amdgcn_wmma_f32_16x16x32_bf16(
                      false, a, false, bm, (short)0, acc, false, false);
        }
        __syncthreads();                // next tile written & current consumed
    }

    if (wave == 8 && M == 0) {          // ones-GEMM column 0 = row sums
#pragma unroll
        for (int r = 0; r < 8; ++r) s_rowsum[hi * 8 + r] = acc[r];
    }
    __syncthreads();

    if (wave < 8) {
        int o = wave * 16 + M;
        size_t ob = ((size_t)bidx * N_ + (size_t)(n0 + hi * 8)) * COUT_ + o;
#pragma unroll
        for (int r = 0; r < 8; ++r) {
            float rs  = fmaxf(s_rowsum[hi * 8 + r], 1e-12f);
            float val = acc[r] / rs;
            val = (val > 0.0f) ? val : 0.01f * val;
            out[ob + (size_t)r * COUT_] = val;
        }
    }
}

// ---------------------------------------------------------------------------
extern "C" void kernel_launch(void* const* d_in, const int* in_sizes, int n_in,
                              void* d_out, int out_size, void* d_ws, size_t ws_size,
                              hipStream_t stream) {
    (void)in_sizes; (void)n_in; (void)out_size; (void)ws_size;
    const float* x      = (const float*)d_in[0];
    const float* coords = (const float*)d_in[1];
    const float* W      = (const float*)d_in[2];
    const float* bias   = (const float*)d_in[3];
    const float* sigma  = (const float*)d_in[4];
    float* out          = (float*)d_out;

    const size_t xb_elems  = (size_t)B_ * N_ * CIN_;     // 2M bf16
    const size_t xpt_elems = (size_t)B_ * COUT_ * N_;    // 2M bf16

    unsigned short* xb  = (unsigned short*)d_ws;                          // 4 MB
    unsigned short* xpT = (unsigned short*)((char*)d_ws + xb_elems * 2);  // 4 MB
    unsigned short* Wb  = (unsigned short*)((char*)d_ws + (xb_elems + xpt_elems) * 2);

    const int xpairs = (int)(xb_elems / 2);
    const int wpairs = (COUT_ * CIN_) / 2;
    cvt_bf16_kernel<<<dim3((xpairs + 255) / 256), dim3(256), 0, stream>>>(
        x, (unsigned*)xb, xpairs);
    cvt_bf16_kernel<<<dim3((wpairs + 255) / 256), dim3(256), 0, stream>>>(
        W, (unsigned*)Wb, wpairs);
    proj_kernel<<<dim3(B_ * N_ / 16), dim3(256), 0, stream>>>(xb, Wb, bias, xpT);
    gconv_kernel<<<dim3(B_ * N_ / 16), dim3(288), 0, stream>>>(coords, xpT, sigma, out);
}
